// DHGAT_40888088657986
// MI455X (gfx1250) — matrix operand order repaired and checked by reading, not compiled, not run on hardware
//
#include <hip/hip_runtime.h>

#define NN 50000
#define EE 800000
#define NHEAD 8
#define BN_EPS 1e-5f

typedef __attribute__((ext_vector_type(16))) __bf16 v16bf;
typedef __attribute__((ext_vector_type(8)))  float  v8f;

union BFrag { unsigned short u[16]; uint4 q[2]; v16bf v; };

__device__ __forceinline__ unsigned short f2bf(float x){
  unsigned int u = __float_as_uint(x);
  u += 0x7FFFu + ((u >> 16) & 1u);        // round-to-nearest-even
  return (unsigned short)(u >> 16);
}
__device__ __forceinline__ float eluf(float x){ return x > 0.f ? x : (__expf(x) - 1.f); }
__device__ __forceinline__ float lrelu(float x){ return x > 0.f ? x : 0.2f * x; }

__device__ __forceinline__ void atomicMaxF(float* addr, float val){
  unsigned int* ua = (unsigned int*)addr;
  unsigned int old = *ua;
  while (__uint_as_float(old) < val){
    unsigned int assumed = old;
    old = atomicCAS(ua, assumed, __float_as_uint(val));
    if (old == assumed) break;
  }
}

// ---------------------------------------------------------------------------
// bf16 WMMA GEMM:  Cout[N x M] = act(A[N x K] * W[K x M] + bias)
// Block: 256 threads = 8 waves; block tile 64 rows x 64 cols.
// Wave tile: 16 rows x 32 cols = 2 accumulators sharing one A fragment.
// K staged 64 wide in LDS (A row-major k-contig, W transposed [col][k]) so
// every WMMA fragment is two 16B ds_load_b128 ops. 4 WMMAs per wave per stage.
// Requires K % 64 == 0 (true for all layers here). act: 0=none,1=relu,2=elu
// ---------------------------------------------------------------------------
__global__ __launch_bounds__(256)
void k_gemm_bf16(const float* __restrict__ A, const float* __restrict__ W,
                 const float* __restrict__ bias, float* __restrict__ Cout,
                 int Nrows, int K, int M, int act)
{
  __shared__ unsigned short As[64 * 72];   // [row][k], stride 144B (16B aligned)
  __shared__ unsigned short Bs[64 * 72];   // [col][k] (transposed W tile)

  const int t    = threadIdx.x;
  const int lane = t & 31;
  const int wave = t >> 5;       // 0..7
  const int twr  = wave & 3;     // wave tile row 0..3  (x16 rows)
  const int twc  = wave >> 2;    // wave tile col 0..1  (x32 cols)
  const int rowBase = blockIdx.x * 64;
  const int colBase = blockIdx.y * 64;

  const int half = lane >> 4;    // 0/1
  const int mloc = lane & 15;

  v8f acc0 = {}, acc1 = {};

  for (int k0 = 0; k0 < K; k0 += 64){
    // ---- stage A tile (64 rows x 64 k) : float4 loads, branch-free guard ----
#pragma unroll
    for (int i = 0; i < 4; ++i){
      int q = i * 256 + t;
      int r = q >> 4;
      int k = (q & 15) * 4;
      int gr = rowBase + r;
      int cr = gr < Nrows ? gr : Nrows - 1;
      const float4 va = *(const float4*)&A[(size_t)cr * K + k0 + k];
      float x0 = va.x, x1 = va.y, x2 = va.z, x3 = va.w;
      if (gr >= Nrows){ x0 = x1 = x2 = x3 = 0.f; }
      uint2 p;
      p.x = (unsigned)f2bf(x0) | ((unsigned)f2bf(x1) << 16);
      p.y = (unsigned)f2bf(x2) | ((unsigned)f2bf(x3) << 16);
      *(uint2*)&As[r * 72 + k] = p;
    }
    // ---- stage W tile transposed (64 cols x 64 k) ----
#pragma unroll
    for (int i = 0; i < 4; ++i){
      int q = i * 256 + t;
      int c = q & 63;            // col fastest -> coalesced global loads
      int k = (q >> 6) * 4;
      int gc = colBase + c;
      int cc = gc < M ? gc : M - 1;
      float x0 = W[(size_t)(k0 + k + 0) * M + cc];
      float x1 = W[(size_t)(k0 + k + 1) * M + cc];
      float x2 = W[(size_t)(k0 + k + 2) * M + cc];
      float x3 = W[(size_t)(k0 + k + 3) * M + cc];
      if (gc >= M){ x0 = x1 = x2 = x3 = 0.f; }
      uint2 p;
      p.x = (unsigned)f2bf(x0) | ((unsigned)f2bf(x1) << 16);
      p.y = (unsigned)f2bf(x2) | ((unsigned)f2bf(x3) << 16);
      *(uint2*)&Bs[c * 72 + k] = p;
    }
    __syncthreads();

    const unsigned short* ap  = &As[(twr * 16 + mloc) * 72];
    const unsigned short* bp0 = &Bs[(twc * 32 + mloc) * 72];
    const unsigned short* bp1 = &Bs[(twc * 32 + 16 + mloc) * 72];
#pragma unroll
    for (int s = 0; s < 2; ++s){
      BFrag af, bf0, bf1;
      // A 16x32 layout: lanes 0-15 K=0..15, lanes 16-31 K=16..31 (contiguous)
      const int ka = s * 32 + half * 16;
      af.q[0] = *(const uint4*)(ap + ka);
      af.q[1] = *(const uint4*)(ap + ka + 8);
      // B 32x16 layout: lanes 0-15 K=0..15, lanes 16-31 K=16..31 (contiguous)
      const int kb = s * 32 + half * 16;
      bf0.q[0] = *(const uint4*)(bp0 + kb);
      bf0.q[1] = *(const uint4*)(bp0 + kb + 8);
      bf1.q[0] = *(const uint4*)(bp1 + kb);
      bf1.q[1] = *(const uint4*)(bp1 + kb + 8);
      acc0 = __builtin_amdgcn_wmma_f32_16x16x32_bf16(false, af.v, false, bf0.v,
                                                     (short)0, acc0, false, false);
      acc1 = __builtin_amdgcn_wmma_f32_16x16x32_bf16(false, af.v, false, bf1.v,
                                                     (short)0, acc1, false, false);
    }
    __syncthreads();
  }

  // C/D layout: VGPR v -> m = v + 8*half, n = lane%16
  const int gc0 = colBase + twc * 32 + mloc;
  const int gc1 = gc0 + 16;
#pragma unroll
  for (int v = 0; v < 8; ++v){
    int gr = rowBase + twr * 16 + v + 8 * half;
    if (gr < Nrows){
      if (gc0 < M){
        float val = acc0[v];
        if (bias) val += bias[gc0];
        if (act == 1) val = val > 0.f ? val : 0.f;
        else if (act == 2) val = eluf(val);
        Cout[(size_t)gr * M + gc0] = val;
      }
      if (gc1 < M){
        float val = acc1[v];
        if (bias) val += bias[gc1];
        if (act == 1) val = val > 0.f ? val : 0.f;
        else if (act == 2) val = eluf(val);
        Cout[(size_t)gr * M + gc1] = val;
      }
    }
  }
}

// ---------------------------------------------------------------------------
// GAT helper kernels
// ---------------------------------------------------------------------------
__global__ void k_fill(float* __restrict__ p, float v, long long n){
  long long i = (long long)blockIdx.x * blockDim.x + threadIdx.x;
  if (i < n) p[i] = v;
}

// per-(node,head) attention logits a_s, a_d
__global__ void k_attn_nh(const float* __restrict__ Hb, const float* __restrict__ asrc,
                          const float* __restrict__ adst, float* __restrict__ AS,
                          float* __restrict__ AD, int C)
{
  int idx = blockIdx.x * blockDim.x + threadIdx.x;
  if (idx >= NN * NHEAD) return;
  int n = idx >> 3, h = idx & 7;
  const float* hp = &Hb[(size_t)n * (NHEAD * C) + h * C];
  float s = 0.f, d = 0.f;
  for (int c = 0; c < C; ++c){
    float v = hp[c];
    s += v * asrc[h * C + c];
    d += v * adst[h * C + c];
  }
  AS[idx] = s; AD[idx] = d;
}

__global__ void k_edge_max(const float* __restrict__ AS, const float* __restrict__ AD,
                           const int* __restrict__ ei, float* __restrict__ MX)
{
  long long idx = (long long)blockIdx.x * blockDim.x + threadIdx.x;
  const long long tot = (long long)(EE + NN) * NHEAD;
  if (idx >= tot) return;
  int e = (int)(idx >> 3), h = (int)(idx & 7);
  int s, d;
  if (e < EE){ s = ei[e]; d = ei[EE + e]; } else { s = d = e - EE; }
  float v = lrelu(AS[s * NHEAD + h] + AD[d * NHEAD + h]);
  atomicMaxF(&MX[d * NHEAD + h], v);
}

__global__ void k_edge_sum(const float* __restrict__ AS, const float* __restrict__ AD,
                           const int* __restrict__ ei, const float* __restrict__ MX,
                           float* __restrict__ SM)
{
  long long idx = (long long)blockIdx.x * blockDim.x + threadIdx.x;
  const long long tot = (long long)(EE + NN) * NHEAD;
  if (idx >= tot) return;
  int e = (int)(idx >> 3), h = (int)(idx & 7);
  int s, d;
  if (e < EE){ s = ei[e]; d = ei[EE + e]; } else { s = d = e - EE; }
  float v = lrelu(AS[s * NHEAD + h] + AD[d * NHEAD + h]);
  atomicAdd(&SM[d * NHEAD + h], __expf(v - MX[d * NHEAD + h]));
}

// scatter h[src]*alpha into out[dst]; one thread per (edge, 4 channels)
__global__ void k_edge_agg(const float* __restrict__ Hb, const float* __restrict__ AS,
                           const float* __restrict__ AD, const float* __restrict__ MX,
                           const float* __restrict__ SM, const int* __restrict__ ei,
                           float* __restrict__ Ob, int C)
{
  const int QC = 2 * C;                 // (8*C)/4 quads per edge
  long long idx = (long long)blockIdx.x * blockDim.x + threadIdx.x;
  const long long tot = (long long)(EE + NN) * QC;
  if (idx >= tot) return;
  int e = (int)(idx / QC);
  int q = (int)(idx % QC);
  int cbase = q * 4;
  int h = cbase / C;
  int s, d;
  if (e < EE){ s = ei[e]; d = ei[EE + e]; } else { s = d = e - EE; }
  float el = lrelu(AS[s * NHEAD + h] + AD[d * NHEAD + h]);
  float alpha = __expf(el - MX[d * NHEAD + h]) / SM[d * NHEAD + h];
  const int HC = NHEAD * C;
  const float* hp = &Hb[(size_t)s * HC + cbase];
  float* op = &Ob[(size_t)d * HC + cbase];
#pragma unroll
  for (int j = 0; j < 4; ++j) atomicAdd(&op[j], hp[j] * alpha);
}

__global__ void k_bias_elu(float* __restrict__ X, const float* __restrict__ b, int F){
  long long idx = (long long)blockIdx.x * blockDim.x + threadIdx.x;
  if (idx >= (long long)NN * F) return;
  int f = (int)(idx % F);
  X[idx] = eluf(X[idx] + b[f]);
}

// ---------------------------------------------------------------------------
// BatchNorm (batch stats, width fixed 256)
// ---------------------------------------------------------------------------
__global__ void k_bn_stats(const float* __restrict__ X, float* __restrict__ mean,
                           float* __restrict__ rstd)
{
  __shared__ float ssum[256], ssq[256];
  int f = blockIdx.x, t = threadIdx.x;
  float sum = 0.f, sq = 0.f;
  for (int n = t; n < NN; n += 256){
    float v = X[(size_t)n * 256 + f];
    sum += v; sq += v * v;
  }
  ssum[t] = sum; ssq[t] = sq; __syncthreads();
  for (int s = 128; s > 0; s >>= 1){
    if (t < s){ ssum[t] += ssum[t + s]; ssq[t] += ssq[t + s]; }
    __syncthreads();
  }
  if (t == 0){
    float mu = ssum[0] / (float)NN;
    float var = ssq[0] / (float)NN - mu * mu;
    mean[f] = mu;
    rstd[f] = rsqrtf(var + BN_EPS);
  }
}

__global__ void k_bn_apply(float* __restrict__ X, const float* __restrict__ mean,
                           const float* __restrict__ rstd, const float* __restrict__ g,
                           const float* __restrict__ b)
{
  long long idx = (long long)blockIdx.x * blockDim.x + threadIdx.x;
  if (idx >= (long long)NN * 256) return;
  int f = (int)(idx & 255);
  X[idx] = (X[idx] - mean[f]) * rstd[f] * g[f] + b[f];
}

// attention gate second linear (128 -> 1) + sigmoid, per node
__global__ void k_gate2(const float* __restrict__ Hmid, const float* __restrict__ w2,
                        const float* __restrict__ b2, float* __restrict__ out)
{
  int n = blockIdx.x * blockDim.x + threadIdx.x;
  if (n >= NN) return;
  const float* x = &Hmid[(size_t)n * 128];
  float a = b2[0];
  for (int k = 0; k < 128; ++k) a += x[k] * w2[k];
  out[n] = 1.f / (1.f + __expf(-a));
}

// x = concat(c*wc, s*ws) with [wc,ws] = softmax([cs,ss])
__global__ void k_concat(const float* __restrict__ C, const float* __restrict__ S,
                         const float* __restrict__ CS, const float* __restrict__ SS,
                         float* __restrict__ X)
{
  long long idx = (long long)blockIdx.x * blockDim.x + threadIdx.x;
  if (idx >= (long long)NN * 512) return;
  int n = (int)(idx >> 9), j = (int)(idx & 511);
  float ec = __expf(CS[n]), es = __expf(SS[n]);
  float inv = 1.f / (ec + es);
  float v = (j < 256) ? C[(size_t)n * 256 + j] * (ec * inv)
                      : S[(size_t)n * 256 + (j - 256)] * (es * inv);
  X[idx] = v;
}

// final 32 -> 2 linear + log_softmax
__global__ void k_logits(const float* __restrict__ X2, const float* __restrict__ w,
                         const float* __restrict__ b, float* __restrict__ out)
{
  int n = blockIdx.x * blockDim.x + threadIdx.x;
  if (n >= NN) return;
  const float* x = &X2[(size_t)n * 32];
  float l0 = b[0], l1 = b[1];
  for (int k = 0; k < 32; ++k){
    l0 += x[k] * w[k * 2 + 0];
    l1 += x[k] * w[k * 2 + 1];
  }
  float m = fmaxf(l0, l1);
  float lse = m + logf(__expf(l0 - m) + __expf(l1 - m));
  out[n * 2 + 0] = l0 - lse;
  out[n * 2 + 1] = l1 - lse;
}

// ---------------------------------------------------------------------------
// Host side
// ---------------------------------------------------------------------------
static inline int cdiv(long long a, long long b){ return (int)((a + b - 1) / b); }

static void launch_gemm(const float* A, const float* W, const float* bias, float* Cout,
                        int K, int M, int act, hipStream_t st)
{
  dim3 g(cdiv(NN, 64), cdiv(M, 64));
  k_gemm_bf16<<<g, 256, 0, st>>>(A, W, bias, Cout, NN, K, M, act);
}

static void gat_layer(const float* X, int K, const float* W, const float* asrc,
                      const float* adst, const float* bias, int C, const int* ei,
                      float* Hb, float* Ob, float* AS, float* AD, float* MX, float* SM,
                      hipStream_t st)
{
  const int HC = NHEAD * C;
  launch_gemm(X, W, nullptr, Hb, K, HC, 0, st);
  k_attn_nh<<<cdiv((long long)NN * NHEAD, 256), 256, 0, st>>>(Hb, asrc, adst, AS, AD, C);
  k_fill<<<cdiv((long long)NN * NHEAD, 256), 256, 0, st>>>(MX, -3.0e38f, (long long)NN * NHEAD);
  k_fill<<<cdiv((long long)NN * NHEAD, 256), 256, 0, st>>>(SM, 0.f, (long long)NN * NHEAD);
  k_fill<<<cdiv((long long)NN * HC, 256), 256, 0, st>>>(Ob, 0.f, (long long)NN * HC);
  const long long eh = (long long)(EE + NN) * NHEAD;
  k_edge_max<<<cdiv(eh, 256), 256, 0, st>>>(AS, AD, ei, MX);
  k_edge_sum<<<cdiv(eh, 256), 256, 0, st>>>(AS, AD, ei, MX, SM);
  const long long at = (long long)(EE + NN) * (2 * C);
  k_edge_agg<<<cdiv(at, 256), 256, 0, st>>>(Hb, AS, AD, MX, SM, ei, Ob, C);
  k_bias_elu<<<cdiv((long long)NN * HC, 256), 256, 0, st>>>(Ob, bias, HC);
}

extern "C" void kernel_launch(void* const* d_in, const int* in_sizes, int n_in,
                              void* d_out, int out_size, void* d_ws, size_t ws_size,
                              hipStream_t stream)
{
  (void)out_size; (void)ws_size;
  const float* content_x = (const float*)d_in[0];
  const float* social_x  = (const float*)d_in[1];
  const int*   cei       = (const int*)d_in[2];
  const int*   sei       = (const int*)d_in[3];

  // param dict may be flattened in insertion order or alphabetically (jax tree).
  // insertion: input 12 is s0.W (65536 elems); alphabetical: input 12 is cab1 (128).
  bool insertion = (n_in > 12) && (in_sizes[12] == 65536);
  static const int alphaMap[34] = {
    0,1,2,3,   4,5,6,7,   20,21,22,23, 24,25,26,27,
    13,12,33,32,          // cbn_g, cbn_b, sbn_g, sbn_b
    10,8,11,9,            // caw1, cab1, caw2, cab2
    30,28,31,29,          // saw1, sab1, saw2, sab2
    15,14,17,16,19,18 };  // f1w, f1b, f2w, f2b, f3w, f3b
  const float* P[34];
  for (int i = 0; i < 34; ++i){
    int a = insertion ? i : alphaMap[i];
    P[i] = (const float*)d_in[4 + a];
  }
  const float *c0W=P[0], *c0as=P[1], *c0ad=P[2], *c0b=P[3];
  const float *c1W=P[4], *c1as=P[5], *c1ad=P[6], *c1b=P[7];
  const float *s0W=P[8], *s0as=P[9], *s0ad=P[10], *s0b=P[11];
  const float *s1W=P[12], *s1as=P[13], *s1ad=P[14], *s1b=P[15];
  const float *cbng=P[16], *cbnb=P[17], *sbng=P[18], *sbnb=P[19];
  const float *caw1=P[20], *cab1=P[21], *caw2=P[22], *cab2=P[23];
  const float *saw1=P[24], *sab1=P[25], *saw2=P[26], *sab2=P[27];
  const float *f1w=P[28], *f1b=P[29], *f2w=P[30], *f2b=P[31], *f3w=P[32], *f3b=P[33];

  // workspace layout (floats)
  float* ws = (float*)d_ws;
  size_t o = 0;
  float* BH  = ws + o; o += (size_t)NN * 512;   // h buffer (pre-aggregation)
  float* BO  = ws + o; o += (size_t)NN * 512;   // aggregated output / activations
  float* BC  = ws + o; o += (size_t)NN * 256;   // content branch result c
  float* BS_ = ws + o; o += (size_t)NN * 256;   // social branch result s
  float* AS  = ws + o; o += (size_t)NN * NHEAD;
  float* AD  = ws + o; o += (size_t)NN * NHEAD;
  float* MX  = ws + o; o += (size_t)NN * NHEAD;
  float* SM  = ws + o; o += (size_t)NN * NHEAD;
  float* STAT= ws + o; o += 1024;               // mean/rstd for c and s
  float* CSb = ws + o; o += (size_t)NN;
  float* SSb = ws + o; o += (size_t)NN;

  // ---- content channel ----
  gat_layer(content_x, 768, c0W, c0as, c0ad, c0b, 64, cei, BH, BO, AS, AD, MX, SM, stream);
  gat_layer(BO,        512, c1W, c1as, c1ad, c1b, 32, cei, BH, BC, AS, AD, MX, SM, stream);
  // ---- social channel ----
  gat_layer(social_x,  256, s0W, s0as, s0ad, s0b, 32, sei, BH, BO, AS, AD, MX, SM, stream);
  gat_layer(BO,        256, s1W, s1as, s1ad, s1b, 32, sei, BH, BS_, AS, AD, MX, SM, stream);

  // ---- batch norm ----
  k_bn_stats<<<256, 256, 0, stream>>>(BC, STAT + 0, STAT + 256);
  k_bn_apply<<<cdiv((long long)NN * 256, 256), 256, 0, stream>>>(BC, STAT + 0, STAT + 256, cbng, cbnb);
  k_bn_stats<<<256, 256, 0, stream>>>(BS_, STAT + 512, STAT + 768);
  k_bn_apply<<<cdiv((long long)NN * 256, 256), 256, 0, stream>>>(BS_, STAT + 512, STAT + 768, sbng, sbnb);

  // ---- channel attention gates ----
  launch_gemm(BC,  caw1, cab1, BH, 256, 128, 1, stream);   // relu
  k_gate2<<<cdiv(NN, 256), 256, 0, stream>>>(BH, caw2, cab2, CSb);
  launch_gemm(BS_, saw1, sab1, BH, 256, 128, 1, stream);
  k_gate2<<<cdiv(NN, 256), 256, 0, stream>>>(BH, saw2, sab2, SSb);

  // ---- fusion ----
  k_concat<<<cdiv((long long)NN * 512, 256), 256, 0, stream>>>(BC, BS_, CSb, SSb, BO);
  launch_gemm(BO, f1w, f1b, BH, 512, 64, 2, stream);       // elu -> x1 in BH
  launch_gemm(BH, f2w, f2b, BC,  64, 32, 2, stream);       // elu -> x2 in BC
  k_logits<<<cdiv(NN, 256), 256, 0, stream>>>(BC, f3w, f3b, (float*)d_out);
}
